// KWSNet_28415503631021
// MI455X (gfx1250) — compile-verified, matrix operand order checked by reading, not run on hardware
//
#include <hip/hip_runtime.h>
#include <hip/hip_bf16.h>

// ---------------------------------------------------------------------------
// KWSNet streaming pipeline for MI455X (gfx1250), wave32 + WMMA bf16.
//   1) prep_weights : fp32 -> bf16 weight conversion (+ conv im2col weight pad)
//   2) conv_gemm    : im2col conv (K'=400 padded to 512) + ReLU -> seq (bf16)
//   3) proj_gemm    : xg = seq @ w_ih^T + b_ih for both directions (fp32 out)
//   4) gru_scan     : 1996-step recurrence, 1 WGP per direction, WMMA bf16,
//                     gate-aligned column mapping -> in-register gate fusion
//   5) clf_dot      : enc @ clf_w^T + clf_b  (project to 2 dims BEFORE mean)
//   6) window_out   : 96-tap sliding mean -> d_out (Tout,1,B,2)
// Workspace use ~1.12 GB (seq 65MB, xg 2x392MB, enc 262MB, weights ~2MB, p 1MB).
// ---------------------------------------------------------------------------

typedef __bf16 bf16;
typedef __attribute__((ext_vector_type(16))) __bf16 v16bf;
typedef __attribute__((ext_vector_type(8)))  float  v8f;

struct U4x2 { uint4 lo, hi; };          // 32B, bit-compatible with v16bf
union AccU  { v8f v; float f[8]; };

// Dims (fixed by the reference setup)
#define BB   64
#define MM   80
#define TT   2000
#define CC   256
#define HH   256
#define KK5  5
#define TC   1996          // TT - KK5 + 1
#define ROWS (TC*BB)       // 127744 = 998 * 128
#define WWIN 96            // window_size - K + 1
#define TOUT 1901          // TT - window_size + 1

// ---- 16x16x32 bf16 fragment load (ISA 7.12.2 layout) ----------------------
// lane: m/n = lane&15, half = lane>>4.  Element e: k = 16*(e>>3) + 8*half + (e&7)
// -> two contiguous 8xbf16 (16B) loads at k0 and k0+16, k0 = kbase + 8*half.
static __device__ __forceinline__ v16bf frag_from(const bf16* p) {
  U4x2 s;
  s.lo = *(const uint4*)(p);
  s.hi = *(const uint4*)(p + 16);
  return __builtin_bit_cast(v16bf, s);
}

static __device__ __forceinline__ v8f wmma_bf16(v16bf a, v16bf b, v8f c) {
  // (neg_a, A, neg_b, B, c_mod, C, reuse_a, reuse_b)
  return __builtin_amdgcn_wmma_f32_16x16x32_bf16(false, a, false, b,
                                                 (short)0, c, false, false);
}

// ---------------------------------------------------------------------------
// 1) weight prep: 4x (768x256) fp32->bf16, conv weights -> bf16 [256][512]
//    (im2col K index m*5+k, padded 400..511 with zeros)
// ---------------------------------------------------------------------------
__global__ __launch_bounds__(256)
void prep_weights(const float* __restrict__ wihf, const float* __restrict__ wihr,
                  const float* __restrict__ whhf, const float* __restrict__ whhr,
                  const float* __restrict__ convw,
                  bf16* __restrict__ o_wihf, bf16* __restrict__ o_wihr,
                  bf16* __restrict__ o_whhf, bf16* __restrict__ o_whhr,
                  bf16* __restrict__ o_cw)
{
  const int N1 = 768 * 256;
  int i = blockIdx.x * 256 + threadIdx.x;
  if (i < 4 * N1) {
    int a = i / N1, j = i - a * N1;
    const float* s = (a == 0) ? wihf : (a == 1) ? wihr : (a == 2) ? whhf : whhr;
    bf16* d       = (a == 0) ? o_wihf : (a == 1) ? o_wihr : (a == 2) ? o_whhf : o_whhr;
    d[j] = (bf16)s[j];
  } else {
    int j = i - 4 * N1;                    // 0 .. 131071
    int c = j >> 9, kk = j & 511;
    o_cw[j] = (bf16)((kk < 400) ? convw[c * 400 + kk] : 0.0f);
  }
}

// ---------------------------------------------------------------------------
// 2) conv as WMMA GEMM: rows = (t,b), K' = 512 (padded), N = 256 channels.
//    Block tile 128x128, 8 waves, per-wave 4(m) x 2(n) 16x16 tiles.
// ---------------------------------------------------------------------------
__global__ __launch_bounds__(256)
void conv_gemm(const float* __restrict__ x, const float* __restrict__ conv_b,
               const bf16* __restrict__ cw, bf16* __restrict__ seq)
{
  __shared__ __align__(16) bf16 As[128 * 128];
  __shared__ __align__(16) bf16 Bs[128 * 128];
  const int tid = threadIdx.x, lane = tid & 31, wave = tid >> 5;
  const int half = lane >> 4, l15 = lane & 15;
  const int m0 = blockIdx.x * 128;            // row block over ROWS
  const int n0 = blockIdx.y * 128;            // channel block over 256
  const int wm = (wave >> 2) * 64, wn = (wave & 3) * 32;

  v8f acc[4][2] = {};
  for (int kc = 0; kc < 4; ++kc) {            // 4 K-chunks of 128
    // A: im2col fill (fp32 x -> bf16), zero pad beyond K=400
    for (int i = tid; i < 128 * 128; i += 256) {
      int r = i >> 7, kk = i & 127;
      int kg = kc * 128 + kk;
      float v = 0.0f;
      if (kg < 400) {
        int m = kg / 5, k = kg - 5 * m;
        int row = m0 + r;
        int t = row >> 6, b = row & 63;       // row = t*64 + b
        v = x[(size_t)(b * MM + m) * TT + t + k];
      }
      As[i] = (bf16)v;
    }
    // B: padded conv weights [c][512], contiguous 16B copies
    {
      uint4* bs4 = (uint4*)Bs;
      const uint4* src = (const uint4*)cw;
      for (int i = tid; i < 2048; i += 256) {
        int r = i >> 4, q = i & 15;
        bs4[i] = src[(size_t)(n0 + r) * 64 + kc * 16 + q];
      }
    }
    __syncthreads();
#pragma unroll
    for (int ks = 0; ks < 4; ++ks) {
      v16bf af[4], bfg[2];
#pragma unroll
      for (int mt = 0; mt < 4; ++mt)
        af[mt] = frag_from(&As[(wm + mt * 16 + l15) * 128 + ks * 32 + half * 8]);
#pragma unroll
      for (int nt = 0; nt < 2; ++nt)
        bfg[nt] = frag_from(&Bs[(wn + nt * 16 + l15) * 128 + ks * 32 + half * 8]);
#pragma unroll
      for (int mt = 0; mt < 4; ++mt)
#pragma unroll
        for (int nt = 0; nt < 2; ++nt)
          acc[mt][nt] = wmma_bf16(af[mt], bfg[nt], acc[mt][nt]);
    }
    __syncthreads();
  }
  // epilogue: +bias, ReLU, store bf16 seq[(t*64+b)][c]
#pragma unroll
  for (int mt = 0; mt < 4; ++mt)
#pragma unroll
    for (int nt = 0; nt < 2; ++nt) {
      AccU u; u.v = acc[mt][nt];
      int col = n0 + wn + nt * 16 + l15;
      float bias = conv_b[col];
#pragma unroll
      for (int r = 0; r < 8; ++r) {
        int row = m0 + wm + mt * 16 + r + 8 * half;
        float v = u.f[r] + bias;
        seq[(size_t)row * CC + col] = (bf16)(v > 0.0f ? v : 0.0f);
      }
    }
}

// ---------------------------------------------------------------------------
// 3) input projections: xg = seq @ w_ih^T + b_ih, rows 127744, K=256, N=768.
//    grid.y = dir(2) x 6 N-blocks. fp32 output (GRU gate precision).
// ---------------------------------------------------------------------------
__global__ __launch_bounds__(256)
void proj_gemm(const bf16* __restrict__ seq,
               const bf16* __restrict__ wih_f, const bf16* __restrict__ wih_r,
               const float* __restrict__ bih_f, const float* __restrict__ bih_r,
               float* __restrict__ xg_f, float* __restrict__ xg_r)
{
  __shared__ __align__(16) bf16 As[128 * 128];
  __shared__ __align__(16) bf16 Bs[128 * 128];
  const int tid = threadIdx.x, lane = tid & 31, wave = tid >> 5;
  const int half = lane >> 4, l15 = lane & 15;
  const int m0 = blockIdx.x * 128;
  const int by = blockIdx.y;
  const int dir = by / 6;
  const int n0 = (by - dir * 6) * 128;
  const bf16*  w    = dir ? wih_r : wih_f;
  const float* bias = dir ? bih_r : bih_f;
  float*       xg   = dir ? xg_r  : xg_f;
  const int wm = (wave >> 2) * 64, wn = (wave & 3) * 32;

  v8f acc[4][2] = {};
  for (int kc = 0; kc < 2; ++kc) {            // K=256 in two 128 chunks
    uint4* as4 = (uint4*)As; uint4* bs4 = (uint4*)Bs;
    const uint4* sa = (const uint4*)seq; const uint4* sb = (const uint4*)w;
    for (int i = tid; i < 2048; i += 256) {
      int r = i >> 4, q = i & 15;
      as4[i] = sa[(size_t)(m0 + r) * 32 + kc * 16 + q];
      bs4[i] = sb[(size_t)(n0 + r) * 32 + kc * 16 + q];
    }
    __syncthreads();
#pragma unroll
    for (int ks = 0; ks < 4; ++ks) {
      v16bf af[4], bfg[2];
#pragma unroll
      for (int mt = 0; mt < 4; ++mt)
        af[mt] = frag_from(&As[(wm + mt * 16 + l15) * 128 + ks * 32 + half * 8]);
#pragma unroll
      for (int nt = 0; nt < 2; ++nt)
        bfg[nt] = frag_from(&Bs[(wn + nt * 16 + l15) * 128 + ks * 32 + half * 8]);
#pragma unroll
      for (int mt = 0; mt < 4; ++mt)
#pragma unroll
        for (int nt = 0; nt < 2; ++nt)
          acc[mt][nt] = wmma_bf16(af[mt], bfg[nt], acc[mt][nt]);
    }
    __syncthreads();
  }
#pragma unroll
  for (int mt = 0; mt < 4; ++mt)
#pragma unroll
    for (int nt = 0; nt < 2; ++nt) {
      AccU u; u.v = acc[mt][nt];
      int col = n0 + wn + nt * 16 + l15;
      float bc = bias[col];
#pragma unroll
      for (int r = 0; r < 8; ++r) {
        int row = m0 + wm + mt * 16 + r + 8 * half;
        xg[(size_t)row * 768 + col] = u.f[r] + bc;
      }
    }
}

// ---------------------------------------------------------------------------
// 4) GRU scan. One block (512 thr = 16 waves) per direction, persistent over
//    all 1996 steps. Wave w owns hidden columns [16w,16w+16): it computes the
//    r/z/n gate tiles at 3H columns {16w, 256+16w, 512+16w} for all 64 batch
//    rows (4 m-tiles). WMMA C/D layout puts N = lane&15 -> all three gates of
//    one (batch, unit) land in the same lane/slot: gate math fully in-register.
//    h: bf16 copy in LDS (matmul input) + fp32 slice in registers (carry).
//    Recurrent weights streamed from L2 each k-step (768 KB resident).
// ---------------------------------------------------------------------------
__global__ __launch_bounds__(512)
void gru_scan(const bf16* __restrict__ whh_f, const bf16* __restrict__ whh_r,
              const float* __restrict__ bhh_f, const float* __restrict__ bhh_r,
              const float* __restrict__ xg_f, const float* __restrict__ xg_r,
              float* __restrict__ enc)
{
  const int dir = blockIdx.x;
  const bf16*  whh = dir ? whh_r : whh_f;
  const float* bhh = dir ? bhh_r : bhh_f;
  const float* xg  = dir ? xg_r  : xg_f;

  const int tid = threadIdx.x, lane = tid & 31, wave = tid >> 5;
  const int half = lane >> 4, l15 = lane & 15;
  const int col = wave * 16 + l15;            // hidden unit owned by this lane

  __shared__ __align__(16) bf16 hl[64 * 256];
  for (int i = tid; i < 64 * 256; i += 512) hl[i] = (bf16)0.0f;

  const float bh0 = bhh[0 * 256 + col];
  const float bh1 = bhh[1 * 256 + col];
  const float bh2 = bhh[2 * 256 + col];
  float h[4][8] = {};                         // fp32 carried hidden slice

  __syncthreads();
  for (int t = 0; t < TC; ++t) {
    v8f acc[3][4] = {};
#pragma unroll
    for (int kk = 0; kk < 8; ++kk) {          // K = 256 = 8 x 32
      v16bf af[4], bw[3];
#pragma unroll
      for (int mt = 0; mt < 4; ++mt)
        af[mt] = frag_from(&hl[(mt * 16 + l15) * 256 + kk * 32 + half * 8]);
#pragma unroll
      for (int g = 0; g < 3; ++g)             // streamed from L2
        bw[g] = frag_from(&whh[(size_t)(g * 256 + col) * 256 + kk * 32 + half * 8]);
#pragma unroll
      for (int g = 0; g < 3; ++g)
#pragma unroll
        for (int mt = 0; mt < 4; ++mt)
          acc[g][mt] = wmma_bf16(af[mt], bw[g], acc[g][mt]);
    }
    __syncthreads();                          // all hl reads done
#pragma unroll
    for (int mt = 0; mt < 4; ++mt) {
      AccU ur, uz, un;
      ur.v = acc[0][mt]; uz.v = acc[1][mt]; un.v = acc[2][mt];
#pragma unroll
      for (int r = 0; r < 8; ++r) {
        int b = mt * 16 + r + 8 * half;
        const float* xr = xg + (size_t)(t * 64 + b) * 768;
        float ir = xr[col], iz = xr[256 + col], inn = xr[512 + col];
        float rg = 1.0f / (1.0f + __expf(-(ir + ur.f[r] + bh0)));
        float zg = 1.0f / (1.0f + __expf(-(iz + uz.f[r] + bh1)));
        float ng = tanhf(inn + rg * (un.f[r] + bh2));
        float hv = (1.0f - zg) * ng + zg * h[mt][r];
        h[mt][r] = hv;
        hl[b * 256 + col] = (bf16)hv;
        enc[(size_t)(t * 64 + b) * 512 + dir * 256 + col] = hv;
      }
    }
    __syncthreads();                          // hl writes visible next step
  }
}

// ---------------------------------------------------------------------------
// 5) classifier projection BEFORE the sliding mean: p[(t,b)][0:2]
// ---------------------------------------------------------------------------
__global__ __launch_bounds__(256)
void clf_dot(const float* __restrict__ enc, const float* __restrict__ clf_w,
             const float* __restrict__ clf_b, float* __restrict__ p)
{
  __shared__ float cw[1024];                  // 2 x 512
  const int tid = threadIdx.x;
  for (int i = tid; i < 1024; i += 256) cw[i] = clf_w[i];
  __syncthreads();
  int row = blockIdx.x * 256 + tid;
  if (row >= ROWS) return;
  const float4* e4 = (const float4*)(enc + (size_t)row * 512);
  float a0 = 0.0f, a1 = 0.0f;
  for (int i = 0; i < 128; ++i) {
    float4 e = e4[i];
    const float* w0 = cw + i * 4;
    const float* w1 = cw + 512 + i * 4;
    a0 += e.x * w0[0] + e.y * w0[1] + e.z * w0[2] + e.w * w0[3];
    a1 += e.x * w1[0] + e.y * w1[1] + e.z * w1[2] + e.w * w1[3];
  }
  p[row * 2 + 0] = a0 + clf_b[0];
  p[row * 2 + 1] = a1 + clf_b[1];
}

// ---------------------------------------------------------------------------
// 6) 96-tap sliding mean -> d_out (Tout,1,B,2); flat idx = t*128 + b*2 + o
// ---------------------------------------------------------------------------
__global__ __launch_bounds__(256)
void window_out(const float* __restrict__ p, float* __restrict__ out)
{
  int idx = blockIdx.x * 256 + threadIdx.x;
  if (idx >= TOUT * BB * 2) return;
  int o = idx & 1;
  int b = (idx >> 1) & 63;
  int t = idx >> 7;
  float s = 0.0f;
  for (int k = 0; k < WWIN; ++k)
    s += p[(size_t)((t + k) * 64 + b) * 2 + o];
  out[idx] = s * (1.0f / WWIN);
}

// ---------------------------------------------------------------------------
extern "C" void kernel_launch(void* const* d_in, const int* in_sizes, int n_in,
                              void* d_out, int out_size, void* d_ws, size_t ws_size,
                              hipStream_t stream) {
  const float* x      = (const float*)d_in[0];
  const float* conv_w = (const float*)d_in[1];
  const float* conv_b = (const float*)d_in[2];
  const float* w_ih_f = (const float*)d_in[3];
  const float* w_hh_f = (const float*)d_in[4];
  const float* b_ih_f = (const float*)d_in[5];
  const float* b_hh_f = (const float*)d_in[6];
  const float* w_ih_r = (const float*)d_in[7];
  const float* w_hh_r = (const float*)d_in[8];
  const float* b_ih_r = (const float*)d_in[9];
  const float* b_hh_r = (const float*)d_in[10];
  const float* clf_w  = (const float*)d_in[11];
  const float* clf_b  = (const float*)d_in[12];
  (void)in_sizes; (void)n_in; (void)out_size; (void)ws_size;

  char* wsb = (char*)d_ws;
  size_t off = 0;
  auto alloc = [&](size_t bytes) -> void* {
    void* pp = wsb + off;
    off += (bytes + 255) & ~(size_t)255;
    return pp;
  };
  bf16*  seq    = (bf16*) alloc((size_t)ROWS * CC * 2);        //  65.4 MB
  bf16*  o_cw   = (bf16*) alloc((size_t)256 * 512 * 2);        //  0.26 MB
  bf16*  o_wihf = (bf16*) alloc((size_t)768 * 256 * 2);
  bf16*  o_wihr = (bf16*) alloc((size_t)768 * 256 * 2);
  bf16*  o_whhf = (bf16*) alloc((size_t)768 * 256 * 2);
  bf16*  o_whhr = (bf16*) alloc((size_t)768 * 256 * 2);
  float* xg_f   = (float*)alloc((size_t)ROWS * 768 * 4);       // 392.4 MB
  float* xg_r   = (float*)alloc((size_t)ROWS * 768 * 4);       // 392.4 MB
  float* enc    = (float*)alloc((size_t)ROWS * 512 * 4);       // 261.6 MB
  float* p      = (float*)alloc((size_t)ROWS * 2 * 4);         //   1.0 MB

  prep_weights<<<3584, 256, 0, stream>>>(w_ih_f, w_ih_r, w_hh_f, w_hh_r, conv_w,
                                         o_wihf, o_wihr, o_whhf, o_whhr, o_cw);
  conv_gemm<<<dim3(998, 2), 256, 0, stream>>>(x, conv_b, o_cw, seq);
  proj_gemm<<<dim3(998, 12), 256, 0, stream>>>(seq, o_wihf, o_wihr,
                                               b_ih_f, b_ih_r, xg_f, xg_r);
  gru_scan<<<2, 512, 0, stream>>>(o_whhf, o_whhr, b_hh_f, b_hh_r,
                                  xg_f, xg_r, enc);
  clf_dot<<<499, 256, 0, stream>>>(enc, clf_w, clf_b, p);
  window_out<<<951, 256, 0, stream>>>(p, (float*)d_out);
}
